// Block_29867202576998
// MI455X (gfx1250) — matrix operand order repaired and checked by reading, not compiled, not run on hardware
//
#include <hip/hip_runtime.h>
#include <hip/hip_bf16.h>
#include <math.h>

// ---------------------------------------------------------------------------
// MLA + MoE transformer block for MI455X (gfx1250, wave32, WMMA).
// Routed-expert dense GEMMs dominate (~412 of ~550 GFLOP) -> all projection GEMMs
// run through v_wmma_f32_16x16x32_bf16 (bf16 in, f32 accumulate).
// GEMM is software-pipelined: register-staged global fetch of tile k+1
// overlaps WMMA on tile k (double-buffered LDS, one barrier per K-step).
// ---------------------------------------------------------------------------

namespace {

constexpr int kB = 2, kT = 1024, kD = 2048, kNH = 16, kDH = 128, kDRH = 64, kDC = 512;
constexpr int kNE = 16, kMOE = 1024, kMS = 2048;           // MOE, MOE*NSH
constexpr int kBT = kB * kT;
constexpr float kEPS = 1e-6f;
constexpr float kATTN_SCALE = 0.07216878364870323f;        // 1/sqrt(DH+DRH)

typedef __bf16 bf16;
typedef __attribute__((ext_vector_type(16))) __bf16 v16bf;
typedef __attribute__((ext_vector_type(8)))  float   v8f;

union FragAB { v16bf v; bf16  e[16]; };
union FragC  { v8f   v; float e[8];  };

// ------------------------------ RMSNorm ------------------------------------
__global__ void rmsnorm_bf16_kernel(const float* __restrict__ x,
                                    const float* __restrict__ w,
                                    bf16* __restrict__ out) {
  __shared__ float red[256];
  const int row = blockIdx.x, tid = threadIdx.x;
  const float* xr = x + (size_t)row * kD;
  float s = 0.f;
  for (int i = tid; i < kD; i += 256) { float v = xr[i]; s += v * v; }
  red[tid] = s;
  __syncthreads();
  for (int off = 128; off > 0; off >>= 1) {
    if (tid < off) red[tid] += red[tid + off];
    __syncthreads();
  }
  const float r = rsqrtf(red[0] / (float)kD + kEPS);
  bf16* o = out + (size_t)row * kD;
  for (int i = tid; i < kD; i += 256) o[i] = (bf16)(w[i] * xr[i] * r);
}

// -------------------- Generic WMMA bf16 GEMM -------------------------------
// C[m,n] (+)= sum_k A[m,k] * W(n,k)   (A bf16 row-major, W fp32)
//   WROWS=true : W(n,k) = W[n*wstride + k]   (rows contiguous in k)
//   WROWS=false: W(n,k) = W[k*wstride + n]   (contiguous in n)
// Requirements: M%64==0, K%64==0; N%64==0 unless NGUARD.
// C addressing: z*cz + (m>>10)*cb + (m&1023)*crow + n  (T==1024 rows per b)
// MODE 0: Cf = acc   1: Cb = bf16(acc)   2: Cf += acc   3: Cf += scale[m]*acc
template <int MODE, bool WROWS, bool NGUARD = false>
__global__ void gemm_wmma(const bf16* __restrict__ A, long long lda, long long az,
                          const float* __restrict__ W, long long wstride,
                          long long wz,
                          float* __restrict__ Cf, bf16* __restrict__ Cb,
                          long long crow, long long cb, long long cz,
                          int M, int N, int K,
                          const float* __restrict__ scale, int sstride) {
  __shared__ bf16 As[2][64][32];
  __shared__ bf16 Ws[2][64][32];   // [n][k]

  const int tid  = threadIdx.x;
  const int wave = tid >> 5, lane = tid & 31;
  const int wr = wave >> 1, wc = wave & 1;
  const int bm = blockIdx.y * 64, bn = blockIdx.x * 64;
  const int z  = blockIdx.z;

  const bf16*  Ab = A + (long long)z * az;
  const float* Wb = W + (long long)z * wz;

  FragC c[2][2];
#pragma unroll
  for (int i = 0; i < 2; ++i)
#pragma unroll
    for (int j = 0; j < 2; ++j)
#pragma unroll
      for (int r = 0; r < 8; ++r) c[i][j].e[r] = 0.f;

  const int l15 = lane & 15;
  const int kbA = (lane < 16) ? 0 : 8;    // A frag: K kbA..+7 and kbA+16..+23
  const int kbB = (lane < 16) ? 0 : 8;    // B frag: K kbB..+7 and kbB+16..+23

  // per-thread tile-load coordinates
  const int amRow = tid >> 2, amOff = (tid & 3) * 8;     // A: 16B per pass
  const int wr0 = tid >> 3, wkq = (tid & 7) * 4;         // WROWS W layout
  const int wkk = tid >> 4, wnq = (tid & 15) * 4;        // !WROWS W layout

  uint4  ra[2];
  float4 rw[4];

  // ---- register-staged global fetch (6 independent b128 loads) -----------
  auto fetch = [&](int k0) {
#pragma unroll
    for (int p = 0; p < 2; ++p)
      ra[p] = *(const uint4*)(Ab + (long long)(bm + p * 32 + amRow) * lda + k0 +
                              amOff);
    if constexpr (WROWS) {
#pragma unroll
      for (int p = 0; p < 4; ++p) {
        const int n = p * 16 + wr0;
        if constexpr (NGUARD) {
          rw[p] = make_float4(0.f, 0.f, 0.f, 0.f);
          if (bn + n < N)
            rw[p] = *(const float4*)(Wb + (long long)(bn + n) * wstride + k0 + wkq);
        } else {
          rw[p] = *(const float4*)(Wb + (long long)(bn + n) * wstride + k0 + wkq);
        }
      }
    } else {
#pragma unroll
      for (int p = 0; p < 4; ++p)
        rw[p] = *(const float4*)(Wb + (long long)(k0 + p * 8 + wkk) * wstride +
                                 bn + wnq);
    }
  };

  // ---- commit staged registers into LDS tile buf -------------------------
  auto commit = [&](int buf) {
#pragma unroll
    for (int p = 0; p < 2; ++p)
      *(uint4*)(&As[buf][p * 32 + amRow][amOff]) = ra[p];
    if constexpr (WROWS) {
#pragma unroll
      for (int p = 0; p < 4; ++p) {
        const int n = p * 16 + wr0;
        Ws[buf][n][wkq]     = (bf16)rw[p].x;
        Ws[buf][n][wkq + 1] = (bf16)rw[p].y;
        Ws[buf][n][wkq + 2] = (bf16)rw[p].z;
        Ws[buf][n][wkq + 3] = (bf16)rw[p].w;
      }
    } else {
#pragma unroll
      for (int p = 0; p < 4; ++p) {
        const int kk = p * 8 + wkk;
        Ws[buf][wnq][kk]     = (bf16)rw[p].x;
        Ws[buf][wnq + 1][kk] = (bf16)rw[p].y;
        Ws[buf][wnq + 2][kk] = (bf16)rw[p].z;
        Ws[buf][wnq + 3][kk] = (bf16)rw[p].w;
      }
    }
  };

  // ---- fragments + 4 WMMAs from LDS tile buf -----------------------------
  auto mma = [&](int buf) {
    FragAB a[2], b[2];
#pragma unroll
    for (int i = 0; i < 2; ++i) {
      const int row = wr * 32 + i * 16 + l15;
#pragma unroll
      for (int e = 0; e < 8; ++e) {
        a[i].e[e]     = As[buf][row][kbA + e];
        a[i].e[8 + e] = As[buf][row][kbA + 16 + e];
      }
    }
#pragma unroll
    for (int j = 0; j < 2; ++j) {
      const int col = wc * 32 + j * 16 + l15;
#pragma unroll
      for (int e = 0; e < 8; ++e) {
        b[j].e[e]     = Ws[buf][col][kbB + e];
        b[j].e[8 + e] = Ws[buf][col][kbB + 16 + e];
      }
    }
#pragma unroll
    for (int i = 0; i < 2; ++i)
#pragma unroll
      for (int j = 0; j < 2; ++j)
        c[i][j].v = __builtin_amdgcn_wmma_f32_16x16x32_bf16(
            false, a[i].v, false, b[j].v, (short)0, c[i][j].v, false, false);
  };

  // ---- software-pipelined main loop (K % 64 == 0 at every call site) -----
  fetch(0);
  for (int k0 = 0; k0 < K; k0 += 64) {
    commit(0);
    fetch(k0 + 32);                     // always valid: K%64==0
    if (k0 + 64 < K) {                  // GL2 prefetch two tiles ahead
      __builtin_prefetch(Ab + (long long)(bm + amRow) * lda + k0 + 64, 0, 1);
      if constexpr (WROWS)
        __builtin_prefetch(Wb + (long long)(bn + wr0) * wstride + k0 + 64, 0, 1);
      else
        __builtin_prefetch(Wb + (long long)(k0 + 64 + wkk) * wstride + bn, 0, 1);
    }
    __syncthreads();
    mma(0);
    commit(1);
    if (k0 + 64 < K) fetch(k0 + 64);
    __syncthreads();
    mma(1);
  }

  const int mb = (lane < 16) ? 0 : 8;
#pragma unroll
  for (int i = 0; i < 2; ++i)
#pragma unroll
    for (int j = 0; j < 2; ++j) {
      const int colg = bn + wc * 32 + j * 16 + l15;
      if (NGUARD && colg >= N) continue;   // only the 16-wide gate GEMM
#pragma unroll
      for (int r = 0; r < 8; ++r) {
        const int rowg = bm + wr * 32 + i * 16 + mb + r;   // M%64==0: no guard
        const long long ci = (long long)z * cz + (long long)(rowg >> 10) * cb +
                             (long long)(rowg & 1023) * crow + colg;
        const float v = c[i][j].e[r];
        if constexpr (MODE == 0) Cf[ci] = v;
        else if constexpr (MODE == 1) Cb[ci] = (bf16)v;
        else if constexpr (MODE == 2) Cf[ci] += v;
        else Cf[ci] += scale[(long long)rowg * sstride] * v;
      }
    }
}

// ------------------------------ RoPE ---------------------------------------
__global__ void rope_q_kernel(bf16* __restrict__ qf, const float* __restrict__ fc) {
  const int id = blockIdx.x * blockDim.x + threadIdx.x;   // B*NH*T*32 threads
  const int j = id & 31, t = (id >> 5) & (kT - 1), bh = id >> 15;
  const size_t base = ((size_t)bh * kT + t) * 192 + 128;
  const float re = (float)qf[base + j], im = (float)qf[base + 32 + j];
  const float fr = fc[(t * 32 + j) * 2], fi = fc[(t * 32 + j) * 2 + 1];
  qf[base + j]      = (bf16)(re * fr - im * fi);
  qf[base + 32 + j] = (bf16)(re * fi + im * fr);
}

__global__ void rope_k_bcast_kernel(const float* __restrict__ krp,
                                    const float* __restrict__ fc,
                                    bf16* __restrict__ kf) {
  const int id = blockIdx.x * blockDim.x + threadIdx.x;   // B*T*32 threads
  const int j = id & 31, t = (id >> 5) & (kT - 1), b = id >> 15;
  const float re = krp[((size_t)b * kT + t) * 64 + j];
  const float im = krp[((size_t)b * kT + t) * 64 + 32 + j];
  const float fr = fc[(t * 32 + j) * 2], fi = fc[(t * 32 + j) * 2 + 1];
  const bf16 rr = (bf16)((re * fr - im * fi) * (1.f / kNH));  // faithful /NH
  const bf16 ri = (bf16)((re * fi + im * fr) * (1.f / kNH));
  for (int h = 0; h < kNH; ++h) {
    const size_t base = ((size_t)(b * kNH + h) * kT + t) * 192 + 128;
    kf[base + j] = rr;
    kf[base + 32 + j] = ri;
  }
}

// ---------------- Attention pass A: per-row softmax stats ------------------
__global__ void attn_stats_kernel(const bf16* __restrict__ qf,
                                  const bf16* __restrict__ kf,
                                  float* __restrict__ stats) {
  __shared__ float qs[4][192];
  const int bh = blockIdx.x;
  const int wv = threadIdx.x >> 5, lane = threadIdx.x & 31;
  const int t  = blockIdx.y * 4 + wv;

  const bf16* q = qf + ((size_t)bh * kT + t) * 192;
  for (int d = lane; d < 192; d += 32) qs[wv][d] = (float)q[d];
  __syncthreads();

  float mx = -3.0e38f, sum = 0.f;
  for (int l0 = 0; l0 <= t; l0 += 32) {
    const int l = l0 + lane;
    if (l <= t) {
      const bf16* kr = kf + ((size_t)bh * kT + l) * 192;
      float acc = 0.f;
      for (int d = 0; d < 192; ++d) acc += qs[wv][d] * (float)kr[d];
      const float s = acc * kATTN_SCALE;
      const float nm = fmaxf(mx, s);
      sum = sum * __expf(mx - nm) + __expf(s - nm);
      mx = nm;
    }
  }
  for (int off = 16; off > 0; off >>= 1) {
    const float om = __shfl_xor(mx, off, 32);
    const float os = __shfl_xor(sum, off, 32);
    const float nm = fmaxf(mx, om);
    sum = sum * __expf(mx - nm) + os * __expf(om - nm);
    mx = nm;
  }
  if (lane == 0) {
    stats[(size_t)bh * kT + t] = mx;
    stats[(size_t)kB * kNH * kT + (size_t)bh * kT + t] = sum;
  }
}

// ------- Attention pass B: out2[b,l,k*NH+h] = sum_t p[t,l] * v[t,k] --------
__global__ void attn_out_kernel(const bf16* __restrict__ qf,
                                const bf16* __restrict__ kf,
                                const float* __restrict__ v,
                                const float* __restrict__ stats,
                                bf16* __restrict__ out2) {
  __shared__ bf16 ks[32][192];
  __shared__ float vs[8][128];
  __shared__ float ps[8][32];

  const int bh = blockIdx.x, l0 = blockIdx.y * 32;
  const int tid = threadIdx.x;
  const int wave = tid >> 5, lane = tid & 31;

  for (int i = tid; i < 32 * 192; i += 256)
    ks[i / 192][i % 192] = kf[((size_t)bh * kT + l0 + i / 192) * 192 + (i % 192)];
  __syncthreads();

  float o[16];
#pragma unroll
  for (int r = 0; r < 16; ++r) o[r] = 0.f;

  for (int t0 = l0; t0 < kT; t0 += 8) {
    __syncthreads();
    for (int i = tid; i < 8 * 128; i += 256)
      vs[i >> 7][i & 127] = v[((size_t)bh * kT + t0 + (i >> 7)) * 128 + (i & 127)];
    {
      const int t = t0 + wave;
      const int l = l0 + lane;
      float p = 0.f;
      if (l <= t) {
        const bf16* qr = qf + ((size_t)bh * kT + t) * 192;
        float acc = 0.f;
        for (int d = 0; d < 192; ++d) acc += (float)qr[d] * (float)ks[lane][d];
        const float m = stats[(size_t)bh * kT + t];
        const float zz = stats[(size_t)kB * kNH * kT + (size_t)bh * kT + t];
        p = __expf(acc * kATTN_SCALE - m) / zz;
      }
      ps[wave][lane] = p;
    }
    __syncthreads();
#pragma unroll
    for (int w = 0; w < 8; ++w) {
      const float pv = ps[w][lane];
      if (pv != 0.f) {
#pragma unroll
        for (int r = 0; r < 16; ++r) o[r] += pv * vs[w][wave * 16 + r];
      }
    }
  }

  const int b = bh >> 4, h = bh & 15;
#pragma unroll
  for (int r = 0; r < 16; ++r) {
    const int k = wave * 16 + r;
    out2[((size_t)b * kT + l0 + lane) * (size_t)kD + (size_t)k * kNH + h] = (bf16)o[r];
  }
}

// ------------------------------ misc elementwise ---------------------------
__global__ void copy_f32_kernel(const float* __restrict__ src, float* __restrict__ dst,
                                int n) {
  const int i = blockIdx.x * blockDim.x + threadIdx.x;
  if (i < n) dst[i] = src[i];
}

__global__ void silu_mul_kernel(const float* __restrict__ a, const float* __restrict__ b,
                                bf16* __restrict__ out, int n) {
  const int i = blockIdx.x * blockDim.x + threadIdx.x;
  if (i >= n) return;
  const float z = a[i] * b[i];
  out[i] = (bf16)(z / (1.f + __expf(-z)));
}

__global__ void routing_kernel(const float* __restrict__ glog,
                               const float* __restrict__ gate_b,
                               float* __restrict__ aff_out,
                               float* __restrict__ rp) {
  const int t = blockIdx.x * blockDim.x + threadIdx.x;
  if (t >= kBT) return;
  float a[kNE];
#pragma unroll
  for (int e = 0; e < kNE; ++e) {
    const float g = glog[(size_t)t * kNE + e];
    a[e] = 1.f / (1.f + __expf(-g)) + gate_b[e];
    aff_out[(size_t)t * kNE + e] = a[e];
  }
  int used = 0;
  float w[4];
  int idx[4];
#pragma unroll
  for (int kk = 0; kk < 4; ++kk) {
    float best = -3.0e38f;
    int bi = 0;
    for (int e = 0; e < kNE; ++e)
      if (!(used & (1 << e)) && a[e] > best) { best = a[e]; bi = e; }
    used |= (1 << bi);
    w[kk] = best;
    idx[kk] = bi;
  }
  const float s = w[0] + w[1] + w[2] + w[3];
#pragma unroll
  for (int e = 0; e < kNE; ++e) rp[(size_t)t * kNE + e] = 0.f;
#pragma unroll
  for (int kk = 0; kk < 4; ++kk) rp[(size_t)t * kNE + idx[kk]] = w[kk] / s;
}

}  // namespace

// ---------------------------------------------------------------------------
extern "C" void kernel_launch(void* const* d_in, const int* in_sizes, int n_in,
                              void* d_out, int out_size, void* d_ws, size_t ws_size,
                              hipStream_t stream) {
  (void)in_sizes; (void)n_in; (void)out_size; (void)ws_size;

  const float* x           = (const float*)d_in[0];
  const float* fc          = (const float*)d_in[1];   // complex64 interleaved (T,32)
  const float* attn_norm_w = (const float*)d_in[3];
  const float* ffn_norm_w  = (const float*)d_in[4];
  const float* gate_w      = (const float*)d_in[5];
  const float* gate_b      = (const float*)d_in[6];
  const float* w1_shared   = (const float*)d_in[7];
  const float* w2_shared   = (const float*)d_in[8];
  const float* w3_shared   = (const float*)d_in[9];
  const float* w1_routed   = (const float*)d_in[10];
  const float* w2_routed   = (const float*)d_in[11];
  const float* w3_routed   = (const float*)d_in[12];
  const float* w_dkv       = (const float*)d_in[13];
  const float* w_uk        = (const float*)d_in[14];
  const float* w_uv        = (const float*)d_in[15];
  const float* w_dq        = (const float*)d_in[16];
  const float* w_uq        = (const float*)d_in[17];
  const float* w_qr        = (const float*)d_in[18];
  const float* w_kr        = (const float*)d_in[19];
  const float* w_o         = (const float*)d_in[20];

  float* out_x = (float*)d_out;                       // (B,T,DIM) residual acc
  float* out_aff = out_x + (size_t)kBT * kD;          // (B,T,NE) affinities

  char* ws = (char*)d_ws;
  bf16*  hb     = (bf16*)(ws + 0);                    //  8 MB  rmsnorm(x) bf16
  bf16*  ckv_b  = (bf16*)(ws + 8388608);              //  2 MB
  bf16*  cq_b   = (bf16*)(ws + 10485760);             //  2 MB
  bf16*  qf     = (bf16*)(ws + 12582912);             // 12 MB (B,NH,T,192)
  bf16*  kf     = (bf16*)(ws + 25165824);             // 12 MB
  float* vbuf   = (float*)(ws + 37748736);            // 16 MB (B,NH,T,128)
  float* krp    = (float*)(ws + 54525952);            // 0.5MB (B,T,64) pre-rope
  float* stats  = (float*)(ws + 55050240);            // 0.25MB max/sum
  bf16*  out2_b = (bf16*)(ws + 55312384);             //  8 MB (B,T,DH*NH)
  bf16*  xn_b   = (bf16*)(ws + 63700992);             //  8 MB
  float* glog   = (float*)(ws + 72089600);            // gate logits
  float* rp     = (float*)(ws + 72220672);            // router probs
  float* s1     = (float*)(ws + 72351744);            // 16 MB
  float* s3     = (float*)(ws + 89128960);            // 16 MB
  bf16*  act_b  = (bf16*)(ws + 105906176);            //  8 MB

  // ---- MLA attention -----------------------------------------------------
  rmsnorm_bf16_kernel<<<kBT, 256, 0, stream>>>(x, attn_norm_w, hb);

  // c_kv, c_q (BTxDC)
  gemm_wmma<1, true><<<dim3(8, 32, 1), 128, 0, stream>>>(hb, kD, 0, w_dkv, kD, 0,
      nullptr, ckv_b, kDC, (long long)kT * kDC, 0, kBT, kDC, kD, nullptr, 0);
  gemm_wmma<1, true><<<dim3(8, 32, 1), 128, 0, stream>>>(hb, kD, 0, w_dq, kD, 0,
      nullptr, cq_b, kDC, (long long)kT * kDC, 0, kBT, kDC, kD, nullptr, 0);

  // per-head up-projections (z = head)
  const long long cbh = (long long)kNH * kT * 192, czh = (long long)kT * 192;
  gemm_wmma<1, true><<<dim3(2, 32, kNH), 128, 0, stream>>>(ckv_b, kDC, 0, w_uk,
      (long long)kNH * kDC, kDC, nullptr, kf, 192, cbh, czh, kBT, kDH, kDC,
      nullptr, 0);
  gemm_wmma<0, true><<<dim3(2, 32, kNH), 128, 0, stream>>>(ckv_b, kDC, 0, w_uv,
      (long long)kNH * kDC, kDC, vbuf, nullptr, kDH,
      (long long)kNH * kT * kDH, (long long)kT * kDH, kBT, kDH, kDC, nullptr, 0);
  gemm_wmma<1, true><<<dim3(2, 32, kNH), 128, 0, stream>>>(cq_b, kDC, 0, w_uq,
      (long long)kNH * kDC, kDC, nullptr, qf, 192, cbh, czh, kBT, kDH, kDC,
      nullptr, 0);
  gemm_wmma<1, true><<<dim3(1, 32, kNH), 128, 0, stream>>>(cq_b, kDC, 0, w_qr,
      (long long)kNH * kDC, kDC, nullptr, qf + 128, 192, cbh, czh, kBT, kDRH,
      kDC, nullptr, 0);
  gemm_wmma<0, true><<<dim3(1, 32, 1), 128, 0, stream>>>(hb, kD, 0, w_kr, kD, 0,
      krp, nullptr, kDRH, (long long)kT * kDRH, 0, kBT, kDRH, kD, nullptr, 0);

  rope_q_kernel<<<(kB * kNH * kT * 32) / 256, 256, 0, stream>>>(qf, fc);
  rope_k_bcast_kernel<<<(kB * kT * 32) / 256, 256, 0, stream>>>(krp, fc, kf);

  attn_stats_kernel<<<dim3(kB * kNH, kT / 4), 128, 0, stream>>>(qf, kf, stats);
  attn_out_kernel<<<dim3(kB * kNH, kT / 32), 256, 0, stream>>>(qf, kf, vbuf,
                                                               stats, out2_b);

  // residual: out = x + attn (w_o projection accumulates in place)
  copy_f32_kernel<<<(kBT * kD) / 256, 256, 0, stream>>>(x, out_x, kBT * kD);
  gemm_wmma<2, true><<<dim3(32, 32, 1), 128, 0, stream>>>(out2_b, kD, 0, w_o,
      (long long)kDH * kNH, 0, out_x, nullptr, kD, (long long)kT * kD, 0,
      kBT, kD, kDH * kNH, nullptr, 0);

  // ---- MoE FFN -----------------------------------------------------------
  rmsnorm_bf16_kernel<<<kBT, 256, 0, stream>>>(out_x, ffn_norm_w, xn_b);

  gemm_wmma<0, true, true><<<dim3(1, 32, 1), 128, 0, stream>>>(xn_b, kD, 0,
      gate_w, kD, 0, glog, nullptr, kNE, (long long)kT * kNE, 0, kBT, kNE, kD,
      nullptr, 0);
  routing_kernel<<<(kBT + 255) / 256, 256, 0, stream>>>(glog, gate_b, out_aff, rp);

  // shared expert
  gemm_wmma<0, false><<<dim3(32, 32, 1), 128, 0, stream>>>(xn_b, kD, 0, w1_shared,
      kMS, 0, s1, nullptr, kMS, (long long)kT * kMS, 0, kBT, kMS, kD, nullptr, 0);
  gemm_wmma<0, false><<<dim3(32, 32, 1), 128, 0, stream>>>(xn_b, kD, 0, w3_shared,
      kMS, 0, s3, nullptr, kMS, (long long)kT * kMS, 0, kBT, kMS, kD, nullptr, 0);
  silu_mul_kernel<<<(kBT * kMS) / 256, 256, 0, stream>>>(s1, s3, act_b, kBT * kMS);
  gemm_wmma<2, false><<<dim3(32, 32, 1), 128, 0, stream>>>(act_b, kMS, 0,
      w2_shared, kD, 0, out_x, nullptr, kD, (long long)kT * kD, 0, kBT, kD, kMS,
      nullptr, 0);

  // routed experts (dense, weighted by router prob in the GEMM epilogue)
  for (int e = 0; e < kNE; ++e) {
    const float* w1e = w1_routed + (size_t)e * kD * kMOE;
    const float* w3e = w3_routed + (size_t)e * kD * kMOE;
    const float* w2e = w2_routed + (size_t)e * kMOE * kD;
    gemm_wmma<0, false><<<dim3(16, 32, 1), 128, 0, stream>>>(xn_b, kD, 0, w1e,
        kMOE, 0, s1, nullptr, kMOE, (long long)kT * kMOE, 0, kBT, kMOE, kD,
        nullptr, 0);
    gemm_wmma<0, false><<<dim3(16, 32, 1), 128, 0, stream>>>(xn_b, kD, 0, w3e,
        kMOE, 0, s3, nullptr, kMOE, (long long)kT * kMOE, 0, kBT, kMOE, kD,
        nullptr, 0);
    silu_mul_kernel<<<(kBT * kMOE) / 256, 256, 0, stream>>>(s1, s3, act_b,
                                                            kBT * kMOE);
    gemm_wmma<3, false><<<dim3(32, 32, 1), 128, 0, stream>>>(act_b, kMOE, 0, w2e,
        kD, 0, out_x, nullptr, kD, (long long)kT * kD, 0, kBT, kD, kMOE, rp + e,
        kNE);
  }
}
